// MultiHeadAttention_48163763257488
// MI455X (gfx1250) — matrix operand order repaired and checked by reading, not compiled, not run on hardware
//
#include <hip/hip_runtime.h>

// ---------------------------------------------------------------------------
// MI455X (gfx1250) multi-head causal attention.
// All matmuls on v_wmma_f32_16x16x32_f16. GEMM tiles are staged by the Tensor
// Data Mover with double-buffered LDS (prefetch tile k+1 under compute of
// tile k, s_wait_tensorcnt<=2). Attention stages K via TDM, V via a manual
// transpose, and runs an online-softmax flash loop on WMMA.
// ---------------------------------------------------------------------------

typedef __attribute__((ext_vector_type(16))) _Float16     v16h;
typedef __attribute__((ext_vector_type(8)))  float        v8f;
typedef __attribute__((ext_vector_type(4)))  unsigned int v4u;
typedef __attribute__((ext_vector_type(8)))  int          v8i_t;
typedef __attribute__((ext_vector_type(4)))  int          v4i_t;

union Frag {
  v16h  v;
  uint4 u[2];
};

#define WMMA_F16(a, b, c) \
  __builtin_amdgcn_wmma_f32_16x16x32_f16(false, (a), false, (b), (short)0, (c), false, false)

#if defined(__AMDGCN__) && __has_builtin(__builtin_amdgcn_tensor_load_to_lds)
#define HAVE_TDM 1
#else
#define HAVE_TDM 0
#endif

static constexpr int Bn  = 4;
static constexpr int Tn  = 2048;
static constexpr int Cn  = 1024;
static constexpr int Hn  = 16;
static constexpr int HSn = 64;

#if HAVE_TDM
// Issue one TDM 2-D tile load: f16 tile [tile_d1 rows][tile_d0 cols] from a
// row-major global matrix with row stride `stride_elems`, into LDS at byte
// offset lds_off (rows packed contiguously, tile_d0*2 bytes apart).
__device__ __forceinline__ void tdm_load_2d(const _Float16* gsrc, unsigned lds_off,
                                            unsigned tile_d0, unsigned tile_d1,
                                            unsigned long long stride_elems)
{
  unsigned long long ga = (unsigned long long)(size_t)gsrc;
  v4u g0;
  g0[0] = 1u;                                   // count = 1 valid descriptor
  g0[1] = lds_off;                              // LDS byte address
  g0[2] = (unsigned)(ga & 0xffffffffu);         // global_addr[31:0]
  g0[3] = (unsigned)((ga >> 32) & 0x01ffffffu)  // global_addr[56:32]
        | (2u << 30);                           // type = 2 ("image")
  v8i_t g1;
  g1[0] = (int)(1u << 16);                      // workgroup_mask=0, data_size=1 (2B)
  g1[1] = (int)(tile_d0 << 16);                 // tensor_dim0[15:0]  (== tile_d0)
  g1[2] = (int)((tile_d0 >> 16) | (tile_d1 << 16));   // tensor_dim0 hi | tensor_dim1 lo
  g1[3] = (int)((tile_d1 >> 16) | (tile_d0 << 16));   // tensor_dim1 hi | tile_dim0
  g1[4] = (int)tile_d1;                         // tile_dim1 (tile_dim2 = 0)
  g1[5] = (int)(unsigned)(stride_elems & 0xffffffffu);        // dim0_stride[31:0]
  g1[6] = (int)(unsigned)((stride_elems >> 32) & 0xffffu);    // dim0_stride[47:32]
  g1[7] = 0;
  v4i_t z4 = {0, 0, 0, 0};
  v8i_t z8 = {0, 0, 0, 0, 0, 0, 0, 0};
  __builtin_amdgcn_tensor_load_to_lds(g0, g1, z4, z4, z8, 0);
}
__device__ __forceinline__ unsigned lds_off_of(const void* p) {
  return (unsigned)(unsigned long long)(size_t)p;   // low 32 bits = LDS byte offset
}
#endif

// ---------------------------------------------------------------------------
// f32 -> f16 conversion kernels
// ---------------------------------------------------------------------------
__global__ void cvt_f16(const float* __restrict__ in, _Float16* __restrict__ out, int n) {
  int i = blockIdx.x * 256 + threadIdx.x;
  if (i < n) out[i] = (_Float16)in[i];
}

// W [H][C][HS] f32  ->  WT [H][HS][C] f16   (so GEMM B-tiles are K-contiguous)
__global__ void cvt_wT(const float* __restrict__ in, _Float16* __restrict__ out) {
  int i   = blockIdx.x * 256 + threadIdx.x;   // over H*C*HS = 1M
  int h   = i >> 16;                          // C*HS = 65536
  int rem = i & 65535;
  int c   = rem >> 6;
  int d   = rem & 63;
  out[((long)(h * HSn + d) << 10) + c] = (_Float16)in[i];
}

// ---------------------------------------------------------------------------
// Tiled WMMA GEMM:  C[m][n] = sum_k A[m][k] * Bm[n][k]   (Bm pre-transposed)
// A: [M][K] f16 row-major.  Bm: [N][K] f16.  Out: f16 [M][N] or f32 + bias.
// BM=128, BN=64, BK=64; 256 threads = 8 waves; wave owns 16 rows x 64 cols.
// Double-buffered TDM staging: prefetch k+1 while computing k.
// ---------------------------------------------------------------------------
template <bool F32OUT>
__global__ __launch_bounds__(256)
void gemm_f16_wmma(const _Float16* __restrict__ A,
                   const _Float16* __restrict__ Bm,
                   void* __restrict__ Cptr,
                   const float* __restrict__ bias,
                   int M, int N, int K)
{
  constexpr int BM = 128, BN = 64, BK = 64;
  __shared__ __align__(16) _Float16 As[2][BM][BK];
  __shared__ __align__(16) _Float16 Bs[2][BN][BK];

  const int  tid  = threadIdx.x;
  const int  lane = tid & 31;
  const int  wid  = tid >> 5;
  const int  lm   = lane & 15;
  const int  hp   = lane >> 4;            // half-wave index
  const int  ae0  = hp ? 8 : 0;           // A-frag K chunk base
  const int  be0  = hp ? 8 : 0;           // B-frag K chunk base
  const long mbase = (long)blockIdx.y * BM;
  const long nbase = (long)blockIdx.x * BN;
  const int  niter = K / BK;

  v8f acc[4] = {};

#if HAVE_TDM
  if (tid == 0) {            // prologue: fill buffer 0
    tdm_load_2d(&A[mbase * K],  lds_off_of(&As[0][0][0]), BK, BM, (unsigned long long)K);
    tdm_load_2d(&Bm[nbase * K], lds_off_of(&Bs[0][0][0]), BK, BN, (unsigned long long)K);
  }
#endif

  for (int i = 0; i < niter; ++i) {
    const int p = i & 1;
#if HAVE_TDM
    __syncthreads();         // prior iteration's reads of buffer p^1 are done
    if (tid == 0) {
      if (i + 1 < niter) {
        tdm_load_2d(&A[mbase * K + (i + 1) * BK],  lds_off_of(&As[p ^ 1][0][0]),
                    BK, BM, (unsigned long long)K);
        tdm_load_2d(&Bm[nbase * K + (i + 1) * BK], lds_off_of(&Bs[p ^ 1][0][0]),
                    BK, BN, (unsigned long long)K);
        __builtin_amdgcn_s_wait_tensorcnt(2);   // buffer p landed; 2 prefetches in flight
      } else {
        __builtin_amdgcn_s_wait_tensorcnt(0);
      }
    }
    __syncthreads();         // publish buffer p to all waves
#else
    __syncthreads();
    {
      const int ch = (tid & 7) * 8;
#pragma unroll
      for (int q = 0; q < 4; ++q) {
        const int row = (tid + q * 256) >> 3;   // 0..127
        *(uint4*)&As[p][row][ch] = *(const uint4*)&A[(mbase + row) * K + i * BK + ch];
      }
#pragma unroll
      for (int q = 0; q < 2; ++q) {
        const int row = (tid + q * 256) >> 3;   // 0..63
        *(uint4*)&Bs[p][row][ch] = *(const uint4*)&Bm[(nbase + row) * K + i * BK + ch];
      }
    }
    __syncthreads();
#endif

#pragma unroll
    for (int ks = 0; ks < 2; ++ks) {
      Frag af;
      const _Float16* ap = &As[p][wid * 16 + lm][ks * 32 + ae0];
      af.u[0] = *(const uint4*)ap;
      af.u[1] = *(const uint4*)(ap + 16);
#pragma unroll
      for (int nt = 0; nt < 4; ++nt) {
        Frag bf;
        const _Float16* bp = &Bs[p][nt * 16 + lm][ks * 32 + be0];
        bf.u[0] = *(const uint4*)bp;
        bf.u[1] = *(const uint4*)(bp + 16);
        acc[nt] = WMMA_F16(af.v, bf.v, acc[nt]);
      }
    }
  }

  // Epilogue: C-layout VGPR r, lane l -> (M = r + 8*(l>=16), N = l%16)
#pragma unroll
  for (int nt = 0; nt < 4; ++nt) {
    const long col = nbase + nt * 16 + lm;
#pragma unroll
    for (int r = 0; r < 8; ++r) {
      const long row = mbase + wid * 16 + r + 8 * hp;
      float val = acc[nt][r];
      if constexpr (F32OUT)
        ((float*)Cptr)[row * N + col] = val + bias[col];
      else
        ((_Float16*)Cptr)[row * N + col] = (_Float16)val;
    }
  }
}

// ---------------------------------------------------------------------------
// Flash attention (causal, online softmax), one (b,h) slice per block-row.
// Block = 256 thr = 8 waves; block covers 128 Q rows; wave owns 16 Q rows.
// q/k/v f16 laid out [B][T][H][HS] (== [M][C], col = h*64+d).
// K tile staged by TDM (overlapped with manual transposed-V staging).
// ---------------------------------------------------------------------------
__global__ __launch_bounds__(256)
void attn_wmma(const _Float16* __restrict__ q,
               const _Float16* __restrict__ k,
               const _Float16* __restrict__ v,
               _Float16* __restrict__ o)
{
  constexpr int QB = 128, KT = 32;
  __shared__ __align__(16) _Float16 Ks[KT][HSn];      // [key][d]
  __shared__ __align__(16) _Float16 Vt[HSn][KT];      // [d][key] (transposed)
  __shared__ __align__(16) _Float16 Pt[8][16][KT];    // per-wave P re-layout scratch

  const int tid  = threadIdx.x;
  const int lane = tid & 31;
  const int wid  = tid >> 5;
  const int lm   = lane & 15;
  const int hp   = lane >> 4;
  const int ae0  = hp ? 8 : 0;
  const int be0  = hp ? 8 : 0;

  const int nqb = Tn / QB;                       // 16
  const int qb  = blockIdx.x % nqb;
  const int h   = (blockIdx.x / nqb) % Hn;
  const int b   = blockIdx.x / (nqb * Hn);
  const int qwbase = qb * QB + wid * 16;         // this wave's first Q row

  // Q fragments (16x64 = two 16x32 A fragments), resident in registers.
  const _Float16* qrow = q + (long)(b * Tn + qwbase + lm) * Cn + h * HSn;
  Frag aq0, aq1;
  aq0.u[0] = *(const uint4*)(qrow + ae0);
  aq0.u[1] = *(const uint4*)(qrow + ae0 + 16);
  aq1.u[0] = *(const uint4*)(qrow + 32 + ae0);
  aq1.u[1] = *(const uint4*)(qrow + 32 + ae0 + 16);

  v8f O[4] = {};
  float mr[8], lr[8], corr[8];
#pragma unroll
  for (int r = 0; r < 8; ++r) { mr[r] = -1e30f; lr[r] = 0.0f; }

  const int kend = (qb + 1) * QB;                // causal horizon for this block
  for (int st = 0; st < kend; st += KT) {
    __syncthreads();
#if HAVE_TDM
    if (tid == 0)
      tdm_load_2d(&k[(long)(b * Tn + st) * Cn + h * HSn], lds_off_of(&Ks[0][0]),
                  HSn, KT, (unsigned long long)Cn);
#else
    {
      const int s  = tid >> 3;
      const int ch = (tid & 7) * 8;
      *(uint4*)&Ks[s][ch] =
          *(const uint4*)&k[(long)(b * Tn + st + s) * Cn + h * HSn + ch];
    }
#endif
    {
      // Stage V transposed [64][32]: read 8 d-contiguous, scatter into Vt.
      const int s2 = tid & 31;
      const int d0 = (tid >> 5) * 8;
      uint4 vv = *(const uint4*)&v[(long)(b * Tn + st + s2) * Cn + h * HSn + d0];
      const _Float16* hv = (const _Float16*)&vv;
#pragma unroll
      for (int j = 0; j < 8; ++j) Vt[d0 + j][s2] = hv[j];
    }
#if HAVE_TDM
    if (tid == 0) __builtin_amdgcn_s_wait_tensorcnt(0);
#endif
    __syncthreads();

    if (st > qwbase + 15) continue;   // tile fully above causal diagonal for this wave

    // S = Q K^T  (16 rows x 32 keys) : two 16x16 C tiles, 2 k-steps each.
    v8f z = {};
    Frag bk;
    const _Float16* kp;
    kp = &Ks[lm][be0];       bk.u[0] = *(const uint4*)kp; bk.u[1] = *(const uint4*)(kp + 16);
    v8f s0 = WMMA_F16(aq0.v, bk.v, z);
    kp = &Ks[lm][32 + be0];  bk.u[0] = *(const uint4*)kp; bk.u[1] = *(const uint4*)(kp + 16);
    s0 = WMMA_F16(aq1.v, bk.v, s0);
    kp = &Ks[16 + lm][be0];      bk.u[0] = *(const uint4*)kp; bk.u[1] = *(const uint4*)(kp + 16);
    v8f s1 = WMMA_F16(aq0.v, bk.v, z);
    kp = &Ks[16 + lm][32 + be0]; bk.u[0] = *(const uint4*)kp; bk.u[1] = *(const uint4*)(kp + 16);
    s1 = WMMA_F16(aq1.v, bk.v, s1);

    // Online softmax. Row M = r + 8*hp lives across the 16 lanes of a half-wave.
#pragma unroll
    for (int r = 0; r < 8; ++r) {
      const int row = qwbase + r + 8 * hp;
      float a0 = s0[r] * 0.125f; if (st + lm > row)      a0 = -1e30f;
      float a1 = s1[r] * 0.125f; if (st + 16 + lm > row) a1 = -1e30f;
      float mx = fmaxf(a0, a1);
      mx = fmaxf(mx, __shfl_xor(mx, 1));
      mx = fmaxf(mx, __shfl_xor(mx, 2));
      mx = fmaxf(mx, __shfl_xor(mx, 4));
      mx = fmaxf(mx, __shfl_xor(mx, 8));
      const float mn = fmaxf(mr[r], mx);
      const float cc = __expf(mr[r] - mn);
      const float p0 = __expf(a0 - mn);
      const float p1 = __expf(a1 - mn);
      float rs = p0 + p1;
      rs += __shfl_xor(rs, 1);
      rs += __shfl_xor(rs, 2);
      rs += __shfl_xor(rs, 4);
      rs += __shfl_xor(rs, 8);
      lr[r] = lr[r] * cc + rs;
      mr[r] = mn;
      corr[r] = cc;
      // Write P to per-wave scratch in [M][k] order for A-frag reload.
      Pt[wid][r + 8 * hp][lm]      = (_Float16)p0;
      Pt[wid][r + 8 * hp][16 + lm] = (_Float16)p1;
    }

#pragma unroll
    for (int dt = 0; dt < 4; ++dt)
#pragma unroll
      for (int r = 0; r < 8; ++r) O[dt][r] *= corr[r];

    // Reload P as a 16x32 A fragment (same-wave LDS store->load, in-order).
    Frag pa;
    const _Float16* pp = &Pt[wid][lm][ae0];
    pa.u[0] = *(const uint4*)pp;
    pa.u[1] = *(const uint4*)(pp + 16);

    // O += P V  : 4 d-tiles of 16.
#pragma unroll
    for (int dt = 0; dt < 4; ++dt) {
      Frag bv;
      const _Float16* vp = &Vt[dt * 16 + lm][be0];
      bv.u[0] = *(const uint4*)vp;
      bv.u[1] = *(const uint4*)(vp + 16);
      O[dt] = WMMA_F16(pa.v, bv.v, O[dt]);
    }
  }

  // Normalize and write attention output [B][T][C], col = h*64 + d.
#pragma unroll
  for (int r = 0; r < 8; ++r) {
    const float inv = 1.0f / lr[r];
    const long  row = (long)(b * Tn + qwbase + r + 8 * hp);
#pragma unroll
    for (int dt = 0; dt < 4; ++dt)
      o[row * Cn + h * HSn + dt * 16 + lm] = (_Float16)(O[dt][r] * inv);
  }
}

// ---------------------------------------------------------------------------
// Launch
// ---------------------------------------------------------------------------
extern "C" void kernel_launch(void* const* d_in, const int* in_sizes, int n_in,
                              void* d_out, int out_size, void* d_ws, size_t ws_size,
                              hipStream_t stream)
{
  const float* x  = (const float*)d_in[0];
  const float* Wq = (const float*)d_in[1];
  const float* Wk = (const float*)d_in[2];
  const float* Wv = (const float*)d_in[3];
  const float* Wo = (const float*)d_in[4];
  const float* bo = (const float*)d_in[5];
  float* out = (float*)d_out;

  char* ws = (char*)d_ws;
  const size_t MB = 1u << 20;
  _Float16* xb  = (_Float16*)(ws);             // [B*T][C]      16 MB
  _Float16* WqT = (_Float16*)(ws + 16 * MB);   // [H][HS][C]     2 MB
  _Float16* WkT = (_Float16*)(ws + 18 * MB);
  _Float16* WvT = (_Float16*)(ws + 20 * MB);
  _Float16* Wob = (_Float16*)(ws + 22 * MB);   // [C][C]         2 MB
  _Float16* qd  = (_Float16*)(ws + 24 * MB);   // [B][T][H][HS] 16 MB
  _Float16* kd  = (_Float16*)(ws + 40 * MB);
  _Float16* vd  = (_Float16*)(ws + 56 * MB);
  _Float16* ao  = (_Float16*)(ws + 72 * MB);   // [B][T][C]     16 MB

  const int M = Bn * Tn;  // 8192

  cvt_f16<<<(Bn * Tn * Cn) / 256, 256, 0, stream>>>(x, xb, Bn * Tn * Cn);
  cvt_f16<<<(Cn * Cn) / 256, 256, 0, stream>>>(Wo, Wob, Cn * Cn);
  cvt_wT<<<(Hn * Cn * HSn) / 256, 256, 0, stream>>>(Wq, WqT);
  cvt_wT<<<(Hn * Cn * HSn) / 256, 256, 0, stream>>>(Wk, WkT);
  cvt_wT<<<(Hn * Cn * HSn) / 256, 256, 0, stream>>>(Wv, WvT);

  dim3 g(Cn / 64, M / 128);  // (16, 64)
  gemm_f16_wmma<false><<<g, 256, 0, stream>>>(xb, WqT, qd, nullptr, M, Cn, Cn);
  gemm_f16_wmma<false><<<g, 256, 0, stream>>>(xb, WkT, kd, nullptr, M, Cn, Cn);
  gemm_f16_wmma<false><<<g, 256, 0, stream>>>(xb, WvT, vd, nullptr, M, Cn, Cn);

  attn_wmma<<<Bn * Hn * (Tn / 128), 256, 0, stream>>>(qd, kd, vd, ao);

  gemm_f16_wmma<true><<<g, 256, 0, stream>>>(ao, Wob, out, bo, M, Cn, Cn);
}
